// HPHiddenPositiveLoss_36730560315728
// MI455X (gfx1250) — matrix operand order repaired and checked by reading, not compiled, not run on hardware
//
#include <hip/hip_runtime.h>
#include <stdint.h>

// ---------------------------------------------------------------------------
// HP hidden-positive contrastive loss — MI455X (gfx1250) WMMA implementation.
//  K1  cvt_bf16        : f32 -> bf16 for z, z_mix, f, s_pr, Pool_ag, Pool_sp
//  K1b gen_base        : threefry2x32 Bernoulli(rho) 784x3136 byte mask
//  K2  rowmax_gemm x2  : Rp = rowmax(f @ Pool_ag^T), Rpe = rowmax(s @ Pool_sp^T)
//                        (A panel staged in LDS via async-to-LDS)
//  K3  fused_mask      : 4 WMMA Gram tiles per 16x16 output tile + mask logic
//  K4  finalize        : 4-group reduction -> scalar loss
// ---------------------------------------------------------------------------

typedef __attribute__((ext_vector_type(16))) __bf16 v16bf;
typedef __attribute__((ext_vector_type(8)))  __bf16 v8bf;
typedef __attribute__((ext_vector_type(8)))  float  v8f;
typedef int hp_v4i __attribute__((vector_size(4 * sizeof(int))));

#define DEVINL __device__ __forceinline__

namespace hp {
constexpr int   B         = 3136;
constexpr int   SPLIT     = 784;
constexpr int   MI        = 4;          // B / SPLIT
constexpr int   PD        = 128;
constexpr int   FD        = 384;
constexpr int   CD        = 512;
constexpr int   POOL      = 16384;
constexpr float TEMP      = 0.6f;
constexpr float BASE_TEMP = 0.07f;
constexpr float RHO       = 0.163f;
constexpr float SCALE     = TEMP / BASE_TEMP;
}

union ABu { v16bf v; v8bf h[2]; };

// ---- async global->LDS staging (CDNA5) ------------------------------------
#if __has_builtin(__builtin_amdgcn_global_load_async_to_lds_b128)
#define HP_ASYNC_LDS 1
#else
#define HP_ASYNC_LDS 0
#endif

DEVINL void hp_wait_async_all() {
#if HP_ASYNC_LDS
#if __has_builtin(__builtin_amdgcn_s_wait_asynccnt)
  __builtin_amdgcn_s_wait_asynccnt(0);
#else
  asm volatile("s_wait_asynccnt 0x0" ::: "memory");
#endif
#endif
}

DEVINL void stage_copy16(const unsigned short* src, unsigned short* dst) {
#if HP_ASYNC_LDS
  __builtin_amdgcn_global_load_async_to_lds_b128(
      (__attribute__((address_space(1))) hp_v4i*)(src),
      (__attribute__((address_space(3))) hp_v4i*)(dst), 0, 0);
#else
  *reinterpret_cast<v8bf*>(dst) = *reinterpret_cast<const v8bf*>(src);
#endif
}

// Copy a contiguous panel (nwords bf16 elements) from global to LDS.
DEVINL void stage_panel(const unsigned short* __restrict__ src,
                        unsigned short* dst, int nwords, int tid, int nthreads) {
  for (int c = tid; c < nwords / 8; c += nthreads)
    stage_copy16(src + c * 8, dst + c * 8);
}

// ---- WMMA tile dot: A panel in LDS, B streamed from global ----------------
// A frag (ISA 7.12.2): lane&15 = M row; lanes<16 hold K {0..7,16..23},
// lanes>=16 hold {8..15,24..31}. B frag: lane&15 = N col; same K split as A
// (lanes<16 K {0..7,16..23}, lanes>=16 K {8..15,24..31}).
// Dual accumulators break the WMMA->WMMA RAW chain (5-NOP hazard, ISA 7.12.1)
// and give the scheduler two independent streams to overlap loads into.
template <int K>
DEVINL v8f wmma_dot_lds(const unsigned short* Ap /*LDS 16xK*/,
                        const unsigned short* __restrict__ Pg,
                        int row16, int col, int lane) {
  static_assert(K % 64 == 0, "K must be a multiple of 64");
  const int kbA = (lane < 16) ? 0 : 8;
  const int kbB = (lane < 16) ? 0 : 8;
  const unsigned short* arow = Ap + row16 * K;
  const unsigned short* brow = Pg + (size_t)col * K + kbB;
  v8f acc0 = {}, acc1 = {};
#pragma unroll
  for (int ko = 0; ko < K; ko += 64) {
    int koA0 = ko + kbA;
    int koA1 = ko + 32 + kbA;
    // Opaque offsets: keep A-fragment reads as in-loop ds_load_b128 instead of
    // letting LICM hoist K/32 fragments into (spilled) VGPRs.
    asm volatile("" : "+v"(koA0));
    asm volatile("" : "+v"(koA1));
    ABu a0, b0, a1, b1;
    a0.h[0] = *reinterpret_cast<const v8bf*>(arow + koA0);
    a0.h[1] = *reinterpret_cast<const v8bf*>(arow + koA0 + 16);
    b0.h[0] = *reinterpret_cast<const v8bf*>(brow + ko);
    b0.h[1] = *reinterpret_cast<const v8bf*>(brow + ko + 16);
    a1.h[0] = *reinterpret_cast<const v8bf*>(arow + koA1);
    a1.h[1] = *reinterpret_cast<const v8bf*>(arow + koA1 + 16);
    b1.h[0] = *reinterpret_cast<const v8bf*>(brow + ko + 32);
    b1.h[1] = *reinterpret_cast<const v8bf*>(brow + ko + 48);
    acc0 = __builtin_amdgcn_wmma_f32_16x16x32_bf16(
        false, a0.v, false, b0.v, (short)0, acc0, false, false);
    acc1 = __builtin_amdgcn_wmma_f32_16x16x32_bf16(
        false, a1.v, false, b1.v, (short)0, acc1, false, false);
  }
  return acc0 + acc1;
}

// ---- K1: f32 -> bf16 (round-to-nearest-even) ------------------------------
__global__ void cvt_bf16_kernel(const float* __restrict__ src,
                                unsigned short* __restrict__ dst, int n) {
  int i = blockIdx.x * blockDim.x + threadIdx.x;
  if (i < n) {
    unsigned u = __float_as_uint(src[i]);
    unsigned r = u + 0x7FFFu + ((u >> 16) & 1u);
    dst[i] = (unsigned short)(r >> 16);
  }
}

// ---- K1b: JAX threefry2x32 Bernoulli(rho) base mask -----------------------
DEVINL unsigned rotl32(unsigned x, int r) { return (x << r) | (x >> (32 - r)); }

DEVINL void threefry2x32(unsigned k0, unsigned k1, unsigned& x0, unsigned& x1) {
  const unsigned ks2 = k0 ^ k1 ^ 0x1BD11BDAu;
  const unsigned ks[3] = {k0, k1, ks2};
  const int rot[8] = {13, 15, 26, 6, 17, 29, 16, 24};
  x0 += k0; x1 += k1;
#pragma unroll
  for (int b = 0; b < 5; ++b) {
#pragma unroll
    for (int r = 0; r < 4; ++r) {
      x0 += x1;
      x1 = rotl32(x1, rot[(b & 1) * 4 + r]);
      x1 ^= x0;
    }
    x0 += ks[(b + 1) % 3];
    x1 += ks[(b + 2) % 3] + (unsigned)(b + 1);
  }
}

__global__ void gen_base_kernel(unsigned char* __restrict__ dst, int n) {
  int i = blockIdx.x * blockDim.x + threadIdx.x;
  if (i >= n) return;
  const unsigned half = (unsigned)n / 2u;   // JAX splits the counter in half
  unsigned x0, x1; int sel;
  if ((unsigned)i < half) { x0 = (unsigned)i;        x1 = (unsigned)i + half; sel = 0; }
  else                    { x0 = (unsigned)i - half; x1 = (unsigned)i;        sel = 1; }
  threefry2x32(0u, 42u, x0, x1);            // jax.random.key(42) -> (0, 42)
  unsigned bits = sel ? x1 : x0;
  float u = __uint_as_float((bits >> 9) | 0x3F800000u) - 1.0f;
  dst[i] = (u < hp::RHO) ? 1 : 0;
}

// ---- K2: GEMM (X @ P^T) with fused row-max epilogue -----------------------
// grid.x = B/16 row blocks; 128 threads = 4 waves, each wave owns a 16-col tile.
template <int K>
__global__ void rowmax_gemm_kernel(const unsigned short* __restrict__ X,
                                   const unsigned short* __restrict__ P,
                                   float* __restrict__ out, int N) {
  __shared__ __align__(16) unsigned short Apanel[16 * K];
  __shared__ float lmax[4][16][16];
  const int tid  = threadIdx.x;
  const int lane = tid & 31;
  const int wave = tid >> 5;
  const int m0   = blockIdx.x * 16;
  const int hi   = (lane >= 16) ? 8 : 0;

  stage_panel(X + (size_t)m0 * K, Apanel, 16 * K, tid, 128);
  hp_wait_async_all();
  __syncthreads();

  float rmax[8];
#pragma unroll
  for (int e = 0; e < 8; ++e) rmax[e] = -3.402823466e38f;

  const int ntiles = N / 64;
  for (int it = 0; it < ntiles; ++it) {
    const int col = (it * 4 + wave) * 16 + (lane & 15);
    if (it + 1 < ntiles)
      __builtin_prefetch((const void*)(P + (size_t)(col + 64) * K), 0, 1);
    v8f acc = wmma_dot_lds<K>(Apanel, P, lane & 15, col, lane);
#pragma unroll
    for (int e = 0; e < 8; ++e) rmax[e] = fmaxf(rmax[e], acc[e]);
  }

#pragma unroll
  for (int e = 0; e < 8; ++e) lmax[wave][e + hi][lane & 15] = rmax[e];
  __syncthreads();
  if (tid < 16) {
    float m = -3.402823466e38f;
    for (int w = 0; w < 4; ++w)
      for (int c = 0; c < 16; ++c) m = fmaxf(m, lmax[w][tid][c]);
    out[m0 + tid] = m;
  }
}

// ---- K3: fused Gram tiles + mask logic + per-row statistics ---------------
// Rows are L2-normalized -> row-max of logits == 1/TEMP (the diagonal), so the
// stabilized logit is (dot - 1)/TEMP and a single pass suffices.
// Epilogue is staged to cap live accumulators at 2: (f,s) dots -> w/negl,
// then the z dot and its stats, then the z_mix dot and its stats.
__global__ void fused_mask_kernel(const unsigned short* __restrict__ zb,
                                  const unsigned short* __restrict__ zmb,
                                  const unsigned short* __restrict__ fb,
                                  const unsigned short* __restrict__ sb,
                                  const float* __restrict__ Rp,
                                  const float* __restrict__ Rpe,
                                  const unsigned char* __restrict__ basep,
                                  const float* __restrict__ lmbd,
                                  float* __restrict__ o_w, float* __restrict__ o_m,
                                  float* __restrict__ o_sz, float* __restrict__ o_sm,
                                  float* __restrict__ o_dz, float* __restrict__ o_dm) {
  __shared__ __align__(16) unsigned short Az [16 * hp::PD];
  __shared__ __align__(16) unsigned short Azm[16 * hp::PD];
  __shared__ __align__(16) unsigned short Af [16 * hp::FD];
  __shared__ __align__(16) unsigned short As [16 * hp::CD];
  __shared__ float lds6[6][4][16][16];

  const float inv_temp = 1.0f / hp::TEMP;
  const int tid  = threadIdx.x;
  const int lane = tid & 31;
  const int wave = tid >> 5;
  const int m0   = blockIdx.x * 16;
  const int hi   = (lane >= 16) ? 8 : 0;
  const float lm = lmbd[0];

  stage_panel(zb  + (size_t)m0 * hp::PD, Az,  16 * hp::PD, tid, 128);
  stage_panel(zmb + (size_t)m0 * hp::PD, Azm, 16 * hp::PD, tid, 128);
  stage_panel(fb  + (size_t)m0 * hp::FD, Af,  16 * hp::FD, tid, 128);
  stage_panel(sb  + (size_t)m0 * hp::CD, As,  16 * hp::CD, tid, 128);
  hp_wait_async_all();
  __syncthreads();

  float Rpi[8], Rpei[8];
  int   ibase[8];
#pragma unroll
  for (int e = 0; e < 8; ++e) {
    const int i = m0 + e + hi;
    Rpi[e]  = Rp[i];
    Rpei[e] = Rpe[i];
    ibase[e] = (i % hp::SPLIT) * hp::B;
  }

  float st[6][8] = {};

  const int ntiles = hp::B / 64;                 // 49
  for (int it = 0; it < ntiles; ++it) {
    const int j0 = (it * 4 + wave) * 16;
    const int j  = j0 + (lane & 15);

    // --- stage 1: threshold masks from the f / s_pr Gram tiles ---
    float wv[8], negl[8];
    {
      v8f aF = wmma_dot_lds<hp::FD>(Af, fb, lane & 15, j, lane);
      v8f aS = wmma_dot_lds<hp::CD>(As, sb, lane & 15, j, lane);
      const float Rpj  = Rp[j];
      const float Rpej = Rpe[j];
#pragma unroll
      for (int e = 0; e < 8; ++e) {
        const int   i    = m0 + e + hi;
        const float neg  = (i == j) ? 0.0f : 1.0f;
        const float Cij  = aF[e];
        const float Ceij = aS[e];
        const float mask = ((Rpi[e]  < Cij)  || (Rpj  < Cij))  ? neg : 0.0f;
        const float mema = ((Rpei[e] < Ceij) || (Rpej < Ceij)) ? neg : 0.0f;
        const bool  nb   = basep[ibase[e] + j] != 0;
        negl[e] = ((mask > 0.0f) || nb) ? neg : 0.0f;
        wv[e]   = mask + mema * lm;
        st[0][e] += wv[e];
        st[1][e] += mask;
      }
    }
    // --- stage 2: z logits ---
    {
      v8f aZ = wmma_dot_lds<hp::PD>(Az, zb, lane & 15, j, lane);
#pragma unroll
      for (int e = 0; e < 8; ++e) {
        const float lz = (aZ[e] - 1.0f) * inv_temp;
        st[2][e] += wv[e] * lz;
        st[4][e] += __expf(lz) * negl[e];
      }
    }
    // --- stage 3: z_mix logits ---
    {
      v8f aM = wmma_dot_lds<hp::PD>(Azm, zmb, lane & 15, j, lane);
#pragma unroll
      for (int e = 0; e < 8; ++e) {
        const float lzm = (aM[e] - 1.0f) * inv_temp;
        st[3][e] += wv[e] * lzm;
        st[5][e] += __expf(lzm) * negl[e];
      }
    }
  }

#pragma unroll
  for (int s = 0; s < 6; ++s)
#pragma unroll
    for (int e = 0; e < 8; ++e) lds6[s][wave][e + hi][lane & 15] = st[s][e];
  __syncthreads();

  if (tid < 96) {
    const int s = tid >> 4;
    const int r = tid & 15;
    float sum = 0.0f;
    for (int w = 0; w < 4; ++w)
      for (int c = 0; c < 16; ++c) sum += lds6[s][w][r][c];
    float* outp[6] = {o_w, o_m, o_sz, o_sm, o_dz, o_dm};
    outp[s][m0 + r] = sum;
  }
}

// ---- K4: group reduction -> scalar loss -----------------------------------
DEVINL float block_sum_1024(float v, float* red) {
  const int t = threadIdx.x;
  red[t] = v;
  __syncthreads();
#pragma unroll
  for (int s = 512; s > 0; s >>= 1) {
    if (t < s) red[t] += red[t + s];
    __syncthreads();
  }
  float r = red[0];
  __syncthreads();
  return r;
}

__global__ void finalize_kernel(const float* __restrict__ wsum,
                                const float* __restrict__ msum,
                                const float* __restrict__ S1z,
                                const float* __restrict__ S1m,
                                const float* __restrict__ Dz,
                                const float* __restrict__ Dm,
                                float* __restrict__ out) {
  __shared__ float red[1024];
  const int t = threadIdx.x;
  float total = 0.0f;
  for (int g = 0; g < hp::MI; ++g) {
    const int  i   = g * hp::SPLIT + t;
    const bool act = t < hp::SPLIT;
    const float ws = act ? wsum[i] : 0.0f;
    const float ms = act ? msum[i] : 0.0f;
    const bool valid = act && (ms > 0.0f);
    const float wv = valid ? ws : 0.0f;

    const float cnt = block_sum_1024(valid ? 1.0f : 0.0f, red);
    const float Sw  = block_sum_1024(wv, red);
    float pnm = wv / fmaxf(Sw, 1e-9f);
    const float psum  = block_sum_1024(pnm, red);
    const float pmean = psum / fmaxf(cnt, 1.0f);
    pnm = pnm / fmaxf(pmean, 1e-9f);

    float termz = 0.0f, termm = 0.0f;
    if (valid) {
      const float mz = (S1z[i] - ws * logf(fmaxf(Dz[i], 1e-9f))) / fmaxf(ws, 1.0f);
      const float mm = (S1m[i] - ws * logf(fmaxf(Dm[i], 1e-9f))) / fmaxf(ws, 1.0f);
      termz = hp::SCALE * mz * pnm;
      termm = hp::SCALE * mm * pnm;
    }
    const float gz = block_sum_1024(termz, red) / fmaxf(cnt, 1.0f);
    const float gm = block_sum_1024(termm, red) / fmaxf(cnt, 1.0f);
    if (cnt > 0.0f) total += gz + gm;
  }
  if (t == 0) out[0] = -total / (float)hp::MI / 2.0f;
}

// ---------------------------------------------------------------------------
extern "C" void kernel_launch(void* const* d_in, const int* in_sizes, int n_in,
                              void* d_out, int out_size, void* d_ws, size_t ws_size,
                              hipStream_t stream) {
  using namespace hp;
  const float* z    = (const float*)d_in[0];
  const float* zm   = (const float*)d_in[1];
  const float* f    = (const float*)d_in[2];
  const float* s    = (const float*)d_in[3];
  const float* pag  = (const float*)d_in[4];
  const float* psp  = (const float*)d_in[5];
  const float* lmbd = (const float*)d_in[6];
  float* out = (float*)d_out;

  char* ws = (char*)d_ws;
  size_t off = 0;
  auto alloc = [&](size_t bytes) -> void* {
    void* p = ws + off;
    off = (off + bytes + 255) & ~(size_t)255;
    return p;
  };

  unsigned short* zb  = (unsigned short*)alloc((size_t)B * PD * 2);
  unsigned short* zmb = (unsigned short*)alloc((size_t)B * PD * 2);
  unsigned short* fb  = (unsigned short*)alloc((size_t)B * FD * 2);
  unsigned short* sb  = (unsigned short*)alloc((size_t)B * CD * 2);
  unsigned short* pab = (unsigned short*)alloc((size_t)POOL * FD * 2);
  unsigned short* psb = (unsigned short*)alloc((size_t)POOL * CD * 2);
  unsigned char*  bas = (unsigned char*) alloc((size_t)SPLIT * B);
  float* Rp   = (float*)alloc((size_t)B * 4);
  float* Rpe  = (float*)alloc((size_t)B * 4);
  float* o_w  = (float*)alloc((size_t)B * 4);
  float* o_m  = (float*)alloc((size_t)B * 4);
  float* o_sz = (float*)alloc((size_t)B * 4);
  float* o_sm = (float*)alloc((size_t)B * 4);
  float* o_dz = (float*)alloc((size_t)B * 4);
  float* o_dm = (float*)alloc((size_t)B * 4);

  auto cvt = [&](const float* src, unsigned short* dst, int n) {
    cvt_bf16_kernel<<<(n + 255) / 256, 256, 0, stream>>>(src, dst, n);
  };
  cvt(z,   zb,  B * PD);
  cvt(zm,  zmb, B * PD);
  cvt(f,   fb,  B * FD);
  cvt(s,   sb,  B * CD);
  cvt(pag, pab, POOL * FD);
  cvt(psp, psb, POOL * CD);

  {
    const int n = SPLIT * B;
    gen_base_kernel<<<(n + 255) / 256, 256, 0, stream>>>(bas, n);
  }

  rowmax_gemm_kernel<FD><<<B / 16, 128, 0, stream>>>(fb, pab, Rp,  POOL);
  rowmax_gemm_kernel<CD><<<B / 16, 128, 0, stream>>>(sb, psb, Rpe, POOL);

  fused_mask_kernel<<<B / 16, 128, 0, stream>>>(zb, zmb, fb, sb, Rp, Rpe, bas,
                                                lmbd, o_w, o_m, o_sz, o_sm,
                                                o_dz, o_dm);

  finalize_kernel<<<1, 1024, 0, stream>>>(o_w, o_m, o_sz, o_sm, o_dz, o_dm, out);
}
